// Yolo_V2_39324720562787
// MI455X (gfx1250) — matrix operand order repaired and checked by reading, not compiled, not run on hardware
//
#include <hip/hip_runtime.h>
#include <math.h>

#define FMP   40
#define HWN   1600
#define KDIM  512
#define NBOX  8000
#define SORTN 8192
#define MROW  256          // u32 words per suppression-mask row (250 used)
#define NTILE 28           // padded 16-wide N tiles: 1 obj + 25 cls + 2 reg
#define WPELEM (NTILE * 16 * 32 * 16)   // 229376 bf16 packed weights

typedef __attribute__((ext_vector_type(16))) __bf16 v16bf;
typedef __attribute__((ext_vector_type(8)))  float  v8f;

// ---------------------------------------------------------------------------
// Kernel 0: pack all three weight matrices into one zero-padded bf16 buffer
// laid out exactly as the WMMA B operand register image:
//   Wp[((nt*16 + ks)*32 + lane)*16 + p] = W_seg[col= n0+(lane&15)]
//                                              [k = ks*32 + p + 16*(lane>>4)]
// so each lane's 16 bf16 (32 B) are contiguous and lane-consecutive (1 KB
// coalesced per wave per k-step). Also packs a padded 448-entry bias vector.
// ---------------------------------------------------------------------------
__global__ __launch_bounds__(256) void pack_weights(
    const float* __restrict__ w_obj, const float* __restrict__ bo,
    const float* __restrict__ w_cls, const float* __restrict__ bc,
    const float* __restrict__ w_reg, const float* __restrict__ br,
    __bf16* __restrict__ Wp, float* __restrict__ biasP)
{
  int t = blockIdx.x * 256 + threadIdx.x;
  if (t < WPELEM) {
    int p    = t & 15;
    int lane = (t >> 4) & 31;
    int ks   = (t >> 9) & 15;
    int nt   = t >> 13;
    const float* Wm; int O, n0;
    if (nt == 0)       { Wm = w_obj; O = 5;   n0 = 0; }
    else if (nt <= 25) { Wm = w_cls; O = 400; n0 = (nt - 1) * 16; }
    else               { Wm = w_reg; O = 20;  n0 = (nt - 26) * 16; }
    int col = n0 + (lane & 15);
    int k = ks * 32 + p + ((lane >> 4) << 4);
    float v = (col < O) ? Wm[col * KDIM + k] : 0.0f;
    Wp[t] = (__bf16)v;
  }
  if (t < NTILE * 16) {
    int nt = t >> 4, n = t & 15;
    const float* Bv; int O, n0;
    if (nt == 0)       { Bv = bo; O = 5;   n0 = 0; }
    else if (nt <= 25) { Bv = bc; O = 400; n0 = (nt - 1) * 16; }
    else               { Bv = br; O = 20;  n0 = (nt - 26) * 16; }
    int col = n0 + n;
    biasP[t] = (col < O) ? Bv[col] : 0.0f;
  }
}

// ---------------------------------------------------------------------------
// Kernel 1: fused head GEMMs (obj/cls/reg) via v_wmma_f32_16x16x32_bf16.
// Each block owns one 16-row (hw) M-tile; stages both 16x512 feature tiles
// in LDS as bf16 in contiguous-k A order (k = 16*half + p); 4 waves sweep 28 N-tiles.
// Inner loop is branch-free: 2x ds_load_b128 (A) + 2x global_load_b128
// (pre-packed bf16 B) + 1 wmma; EXEC stays all-ones.
// ---------------------------------------------------------------------------
__global__ __launch_bounds__(128) void head_gemm(
    const float* __restrict__ cls_feat, const float* __restrict__ reg_feat,
    const __bf16* __restrict__ Wp, const float* __restrict__ biasP,
    float* __restrict__ obj_o, float* __restrict__ cls_o, float* __restrict__ reg_o)
{
  __shared__ __align__(32) __bf16 Alds[2][16][32][16];   // [feat][kstep][lane][p]
  const int hw0 = blockIdx.x * 16;
  const int t   = threadIdx.x;

  // Phase 1: cooperative coalesced load of both 16x512 feature tiles -> LDS bf16
  // A 16x32 layout inverse: lane = m + 16*half; half = kk>>4; p = kk&15
  for (int f = 0; f < 2; ++f) {
    const float* feat = f ? reg_feat : cls_feat;
    #pragma unroll 4
    for (int i = 0; i < 64; ++i) {
      int e = t + i * 128;              // 0..8191
      int m = e & 15;
      int k = e >> 4;
      float v = feat[k * HWN + hw0 + m];
      int kk = k & 31, ks = k >> 5;
      int half = kk >> 4;
      int p = kk & 15;
      Alds[f][ks][m + (half << 4)][p] = (__bf16)v;
    }
  }
  __syncthreads();

  const int wave = t >> 5, lane = t & 31;
  const int n = lane & 15, half = lane >> 4;

  for (int nt = wave; nt < NTILE; nt += 4) {
    int O, n0, fsel, ostride; float* out;
    if (nt == 0)       { O = 5;   n0 = 0;              fsel = 1; out = obj_o; ostride = 5;   }
    else if (nt <= 25) { O = 400; n0 = (nt - 1) * 16;  fsel = 0; out = cls_o; ostride = 400; }
    else               { O = 20;  n0 = (nt - 26) * 16; fsel = 1; out = reg_o; ostride = 20;  }

    const __bf16* wp = Wp + ((size_t)(nt * 16) * 32 + lane) * 16;

    v8f acc = {};
    for (int ks = 0; ks < 16; ++ks) {
      v16bf a = *(const v16bf*)&Alds[fsel][ks][lane][0];
      v16bf b = *(const v16bf*)(wp + (size_t)ks * 32 * 16);
      acc = __builtin_amdgcn_wmma_f32_16x16x32_bf16(
          /*neg_a=*/false, a, /*neg_b=*/false, b,
          /*c_mod=*/(short)0, acc, /*reuse_a=*/false, /*reuse_b=*/false);
    }

    const int col = n0 + n;
    if (col < O) {
      float bias = biasP[nt * 16 + n];
      #pragma unroll
      for (int r = 0; r < 8; ++r) {
        int m = r + half * 8;                     // D layout: VGPR r, lane halves
        out[(hw0 + m) * ostride + col] = acc[r] + bias;
      }
    }
  }
}

// ---------------------------------------------------------------------------
// Kernel 2: per-box sigmoid/score/argmax-80, box decode, NMS-box (cxcywh
// reinterpretation), writes bboxes/score/labels straight to d_out.
// ---------------------------------------------------------------------------
__global__ __launch_bounds__(256) void decode_kernel(
    const float* __restrict__ obj_ws, const float* __restrict__ cls_ws, const float* __restrict__ reg_ws,
    float* __restrict__ out, float* __restrict__ b2, float* __restrict__ scoreU, int* __restrict__ labelU)
{
  int bId = blockIdx.x * 256 + threadIdx.x;
  if (bId >= NBOX) return;
  int hw = bId / 5, a = bId % 5;
  const float AW[5] = {17.f, 55.f, 92.f, 202.f, 289.f};
  const float AH[5] = {25.f, 75.f, 206.f, 21.f, 311.f};

  float o = obj_ws[hw * 5 + a];
  const float* c = cls_ws + hw * 400 + a * 80;
  float best = c[0]; int bl = 0;
  for (int j = 1; j < 80; ++j) { float v = c[j]; if (v > best) { best = v; bl = j; } }

  float score = sqrtf((1.f / (1.f + expf(-o))) * (1.f / (1.f + expf(-best))));

  float tx = reg_ws[hw * 20 + a * 4 + 0];
  float ty = reg_ws[hw * 20 + a * 4 + 1];
  float tw = reg_ws[hw * 20 + a * 4 + 2];
  float th = reg_ws[hw * 20 + a * 4 + 3];
  float ax = (float)(hw % FMP), ay = (float)(hw / FMP);
  float cx = (1.f / (1.f + expf(-tx)) + ax) * 32.f;
  float cy = (1.f / (1.f + expf(-ty)) + ay) * 32.f;
  float w = expf(tw) * AW[a];
  float h = expf(th) * AH[a];
  float x1 = cx - 0.5f * w, y1 = cy - 0.5f * h, x2 = cx + 0.5f * w, y2 = cy + 0.5f * h;

  out[bId * 4 + 0] = x1; out[bId * 4 + 1] = y1;
  out[bId * 4 + 2] = x2; out[bId * 4 + 3] = y2;
  out[4 * NBOX + bId] = score;
  out[5 * NBOX + bId] = (float)bl;

  // faithful re-interpretation of x1y1x2y2 as cxcywh for IoU
  b2[bId * 4 + 0] = x1 - 0.5f * x2;
  b2[bId * 4 + 1] = y1 - 0.5f * y2;
  b2[bId * 4 + 2] = x1 + 0.5f * x2;
  b2[bId * 4 + 3] = y1 + 0.5f * y2;
  scoreU[bId] = score;
  labelU[bId] = bl;
}

// ---------------------------------------------------------------------------
// Kernel 3: bitonic sort (descending score, ascending index tie-break to
// match stable jnp.argsort(-score)) over 8192 padded entries, one workgroup.
// ---------------------------------------------------------------------------
__global__ __launch_bounds__(1024) void sort_kernel(const float* __restrict__ scoreU,
                                                    int* __restrict__ order)
{
  __shared__ float key[SORTN];
  __shared__ int   idx[SORTN];
  int t = threadIdx.x;
  for (int i = t; i < SORTN; i += 1024) {
    key[i] = (i < NBOX) ? scoreU[i] : -3.0e38f;
    idx[i] = i;
  }
  __syncthreads();
  for (int k = 2; k <= SORTN; k <<= 1) {
    for (int j = k >> 1; j > 0; j >>= 1) {
      for (int i = t; i < SORTN; i += 1024) {
        int ixj = i ^ j;
        if (ixj > i) {
          bool up = ((i & k) == 0);
          float ka = key[i], kb = key[ixj];
          int ia = idx[i], ib = idx[ixj];
          bool prec = (ka > kb) || (ka == kb && ia < ib);
          bool doSwap = up ? (!prec) : prec;
          if (doSwap) { key[i] = kb; key[ixj] = ka; idx[i] = ib; idx[ixj] = ia; }
        }
      }
      __syncthreads();
    }
  }
  for (int i = t; i < NBOX; i += 1024) order[i] = idx[i];
}

// ---------------------------------------------------------------------------
// Kernel 4: gather boxes/scores/labels into sorted order, precompute areas.
// ---------------------------------------------------------------------------
__global__ __launch_bounds__(256) void gather_kernel(const int* __restrict__ order,
    const float* __restrict__ b2, const float* __restrict__ scoreU, const int* __restrict__ labelU,
    float* __restrict__ b2s, float* __restrict__ scoreS, int* __restrict__ labelS, float* __restrict__ areaS)
{
  int j = blockIdx.x * 256 + threadIdx.x;
  if (j >= NBOX) return;
  int o = order[j];
  float x1 = b2[o * 4 + 0], y1 = b2[o * 4 + 1], x2 = b2[o * 4 + 2], y2 = b2[o * 4 + 3];
  b2s[j * 4 + 0] = x1; b2s[j * 4 + 1] = y1; b2s[j * 4 + 2] = x2; b2s[j * 4 + 3] = y2;
  areaS[j] = (x2 - x1) * (y2 - y1);
  scoreS[j] = scoreU[o];
  labelS[j] = labelU[o];
}

// ---------------------------------------------------------------------------
// Kernel 5: parallel suppression-mask build. Thread (k,w) computes a 32-bit
// word of "k would suppress j" bits (iou>0.5, same class, j>k). 64M IoUs,
// fully parallel; masks (8 MB) stay in the 192 MB L2.
// ---------------------------------------------------------------------------
__global__ __launch_bounds__(256) void mask_kernel(const float* __restrict__ b2s,
    const float* __restrict__ areaS, const int* __restrict__ labelS, unsigned* __restrict__ M)
{
  int t = blockIdx.x * 256 + threadIdx.x;
  if (t >= NBOX * 250) return;
  int k = t / 250, w = t % 250;
  const float4* bb = (const float4*)b2s;
  float4 bk = bb[k];
  float  ak = areaS[k];
  int    lk = labelS[k];
  unsigned bits = 0u;
  #pragma unroll 4
  for (int b = 0; b < 32; ++b) {
    int j = w * 32 + b;                      // 250*32 == 8000, always in range
    float4 bj = bb[j];
    float xx1 = fmaxf(bk.x, bj.x);
    float yy1 = fmaxf(bk.y, bj.y);
    float xx2 = fminf(bk.z, bj.z);
    float yy2 = fminf(bk.w, bj.w);
    float inter = fmaxf(1e-10f, xx2 - xx1) * fmaxf(1e-10f, yy2 - yy1);
    float iou = inter / (ak + areaS[j] - inter + 1e-14f);
    if (iou > 0.5f && labelS[j] == lk && j > k) bits |= (1u << b);
  }
  M[k * MROW + w] = bits;
}

// ---------------------------------------------------------------------------
// Kernel 6: single-wave sequential greedy scan (exactly the fori_loop
// semantics). Single-wave workgroup => barriers lower to s_nop; the scan is
// L2 mask-row reads + LDS bitmap ORs. Scatters keep back to original order.
// ---------------------------------------------------------------------------
__global__ __launch_bounds__(32) void nms_seq(const unsigned* __restrict__ M,
    const float* __restrict__ scoreS, const int* __restrict__ order, float* __restrict__ out)
{
  __shared__ unsigned supp[MROW];
  int lane = threadIdx.x;
  for (int w = lane; w < MROW; w += 32) supp[w] = 0u;
  __syncthreads();
  for (int k = 0; k < NBOX; ++k) {
    unsigned sw = supp[k >> 5];
    bool sk = (sw >> (k & 31)) & 1u;
    bool vk = scoreS[k] >= 0.3f;
    if (vk && !sk) {
      const unsigned* row = M + k * MROW;
      #pragma unroll
      for (int s = 0; s < 8; ++s) {
        int w = lane + s * 32;
        if (w < 250) supp[w] |= row[w];
      }
    }
    __syncthreads();     // single wave: cheap, just orders LDS traffic
  }
  for (int k = lane; k < NBOX; k += 32) {
    bool sk = (supp[k >> 5] >> (k & 31)) & 1u;
    bool vk = scoreS[k] >= 0.3f;
    out[6 * NBOX + order[k]] = (vk && !sk) ? 1.0f : 0.0f;
  }
}

// ---------------------------------------------------------------------------
extern "C" void kernel_launch(void* const* d_in, const int* in_sizes, int n_in,
                              void* d_out, int out_size, void* d_ws, size_t ws_size,
                              hipStream_t stream)
{
  (void)in_sizes; (void)n_in; (void)out_size; (void)ws_size;
  const float* cls_feat = (const float*)d_in[0];
  const float* reg_feat = (const float*)d_in[1];
  const float* w_obj    = (const float*)d_in[2];
  const float* b_obj    = (const float*)d_in[3];
  const float* w_cls    = (const float*)d_in[4];
  const float* b_cls    = (const float*)d_in[5];
  const float* w_reg    = (const float*)d_in[6];
  const float* b_reg    = (const float*)d_in[7];
  float* out = (float*)d_out;
  float* ws  = (float*)d_ws;

  // workspace layout (float offsets)
  float*  obj_ws = ws + 0;                      //  8000
  float*  reg_ws = ws + 8000;                   // 32000
  float*  cls_ws = ws + 40000;                  // 640000
  float*  b2     = ws + 680000;                 // 32000
  float*  scoreU = ws + 712000;                 //  8000
  int*    labelU = (int*)(ws + 720000);         //  8000
  int*    order  = (int*)(ws + 728000);         //  8000
  float*  b2s    = ws + 736000;                 // 32000
  float*  scoreS = ws + 768000;                 //  8000
  int*    labelS = (int*)(ws + 776000);         //  8000
  float*  areaS  = ws + 784000;                 //  8000
  __bf16* Wp     = (__bf16*)(ws + 792000);      // 229376 bf16 = 114688 floats
  float*  biasP  = ws + 906688;                 //   448
  unsigned* M    = (unsigned*)(ws + 907264);    // 8000*256 u32 = 8 MB

  pack_weights<<<WPELEM / 256, 256, 0, stream>>>(w_obj, b_obj, w_cls, b_cls,
                                                 w_reg, b_reg, Wp, biasP);
  head_gemm<<<100, 128, 0, stream>>>(cls_feat, reg_feat, Wp, biasP,
                                     obj_ws, cls_ws, reg_ws);
  decode_kernel<<<(NBOX + 255) / 256, 256, 0, stream>>>(obj_ws, cls_ws, reg_ws,
                                                        out, b2, scoreU, labelU);
  sort_kernel<<<1, 1024, 0, stream>>>(scoreU, order);
  gather_kernel<<<(NBOX + 255) / 256, 256, 0, stream>>>(order, b2, scoreU, labelU,
                                                        b2s, scoreS, labelS, areaS);
  mask_kernel<<<(NBOX * 250 + 255) / 256, 256, 0, stream>>>(b2s, areaS, labelS, M);
  nms_seq<<<1, 32, 0, stream>>>(M, scoreS, order, out);
}
